// CTRGCN_3702261809371
// MI455X (gfx1250) — matrix operand order repaired and checked by reading, not compiled, not run on hardware
//
#include <hip/hip_runtime.h>
#include <hip/hip_bf16.h>

// ---------------- problem constants ----------------
#define N_    128
#define CIN   64
#define COUT  64
#define T_    256
#define V_    25
#define R_    8
#define TT    16              // t-rows per block in pass 3
#define COLS  (TT * V_)       // 400 columns per x-tile (t,v flattened)
#define NCT   (COLS / 16)     // 25 WMMA N-tiles
#define XB_STRIDE 72          // bf16 per Xb row (64 padded -> 144B, 16B aligned, bank-spread)
#define VP    32              // padded V for WMMA K / N dims

typedef __attribute__((ext_vector_type(16))) __bf16 v16bf;
typedef __attribute__((ext_vector_type(8)))  __bf16 bf16x8;
typedef __attribute__((ext_vector_type(8)))  float  v8f;
typedef __attribute__((ext_vector_type(2)))  float  f32x2;

// ---------------- pass 1: s[n,c,v] = sum_t x[n,c,t,v] ----------------
__global__ void __launch_bounds__(256) k_sum_t(const float* __restrict__ x,
                                               float* __restrict__ s) {
  int nc = blockIdx.x;                       // n*CIN + c
  const float* xp = x + (size_t)nc * T_ * V_;
  int lane = threadIdx.x & 31;
  int grp  = threadIdx.x >> 5;               // 8 t-groups
  float acc = 0.f;
  if (lane < V_) {
    for (int t = grp; t < T_; t += 8) acc += xp[t * V_ + lane];
  }
  __shared__ float red[8][32];
  red[grp][lane] = acc;
  __syncthreads();
  if (threadIdx.x < 32) {
    float tot = 0.f;
    for (int g = 0; g < 8; ++g) tot += red[g][threadIdx.x];
    if (threadIdx.x < V_) s[(size_t)nc * V_ + threadIdx.x] = tot;
  }
}

// ---------------- pass 2: adjb[n,o,u,v] (bf16, padded 32x32) ----------------
__global__ void __launch_bounds__(256) k_adj(const float* __restrict__ s,
                                             const float* __restrict__ A,
                                             const float* __restrict__ W1,
                                             const float* __restrict__ b1,
                                             const float* __restrict__ W2,
                                             const float* __restrict__ b2,
                                             const float* __restrict__ W4,
                                             const float* __restrict__ b4,
                                             __bf16* __restrict__ adjb) {
  int n = blockIdx.x;
  __shared__ float sl[CIN][V_];              // 6.4 KB
  __shared__ float x1[R_][V_], x2[R_][V_];   // 1.6 KB
  __shared__ float y[R_][V_][V_];            // 20 KB
  const float* sp = s + (size_t)n * CIN * V_;
  for (int i = threadIdx.x; i < CIN * V_; i += 256)
    sl[i / V_][i % V_] = sp[i];
  __syncthreads();
  for (int i = threadIdx.x; i < 2 * R_ * V_; i += 256) {
    int which = i / (R_ * V_);
    int rv = i % (R_ * V_);
    int r = rv / V_, v = rv % V_;
    const float* W = which ? W2 : W1;
    float acc = 0.f;
    for (int c = 0; c < CIN; ++c) acc += W[r * CIN + c] * sl[c][v];
    acc = acc * (1.0f / (float)T_) + (which ? b2[r] : b1[r]);
    (which ? x2 : x1)[r][v] = acc;
  }
  __syncthreads();
  for (int i = threadIdx.x; i < R_ * V_ * V_; i += 256) {
    int r = i / (V_ * V_);
    int uv = i % (V_ * V_);
    int u = uv / V_, v = uv % V_;
    y[r][u][v] = tanhf(x1[r][u] - x2[r][v]);
  }
  __syncthreads();
  __bf16* ap = adjb + (size_t)n * COUT * VP * VP;
  for (int i = threadIdx.x; i < COUT * VP * VP; i += 256) {
    int o = i / (VP * VP);
    int uv = i % (VP * VP);
    int u = uv / VP, v = uv % VP;
    float val = 0.f;
    if (u < V_ && v < V_) {
      val = b4[o] + A[u * V_ + v];                 // ALPHA == 1.0 folded
      #pragma unroll
      for (int r = 0; r < R_; ++r) val += W4[o * R_ + r] * y[r][u][v];
    }
    ap[i] = (__bf16)val;
  }
}

// ---------------- pass 3: fused x3 = W3@x (+b3), z = adj (.) x3 ----------------
// X3 LDS region holds x3 directly in WMMA A-fragment coordinates:
//   X3frag[o][lane][e]  where for value (t,v):
//     lane = t + 16*(v>>4),  e = v&15
// so GEMM2's A-operand is one contiguous 32B per-lane LDS read.
__global__ void __launch_bounds__(256, 1)
k_main(const float* __restrict__ x,
       const float* __restrict__ W3,
       const float* __restrict__ b3,
       const __bf16* __restrict__ adjb,
       float* __restrict__ z) {
  extern __shared__ char lds[];
  __bf16* Xb = (__bf16*)lds;                                  // [COLS][XB_STRIDE] 57.6 KB
  __bf16* X3 = (__bf16*)(lds + (size_t)COLS * XB_STRIDE * 2); // [COUT][32][16]    64  KB

  int blk = blockIdx.x;
  int n   = blk / (T_ / TT);
  int tb  = blk % (T_ / TT);
  int t0  = tb * TT;

  const float* xn = x + (size_t)n * CIN * T_ * V_;

  // zero X3 (provides K-padding v=25..31 for GEMM2, in fragment coordinates)
  uint32_t* X3w = (uint32_t*)X3;
  for (int i = threadIdx.x; i < COUT * TT * VP / 2; i += 256) X3w[i] = 0u;

  // stage x-tile transposed: Xb[col][c] = bf16(x[n][c][t0 + col/25][col%25])
  for (int task = threadIdx.x; task < CIN * (COLS / 4); task += 256) {
    int c    = task / (COLS / 4);
    int col4 = (task % (COLS / 4)) * 4;
    float4 f4 = *(const float4*)(xn + (size_t)c * (T_ * V_) + t0 * V_ + col4);
    Xb[(col4 + 0) * XB_STRIDE + c] = (__bf16)f4.x;
    Xb[(col4 + 1) * XB_STRIDE + c] = (__bf16)f4.y;
    Xb[(col4 + 2) * XB_STRIDE + c] = (__bf16)f4.z;
    Xb[(col4 + 3) * XB_STRIDE + c] = (__bf16)f4.w;
  }
  __syncthreads();

  int wave  = threadIdx.x >> 5;
  int lane  = threadIdx.x & 31;
  int lhalf = lane >> 4;          // lane group 0: lanes 0-15, 1: lanes 16-31
  int l16   = lane & 15;

  // ---- GEMM1: x3[o][col] = W3[o][c] * x[c][col], K = 64 (2 steps of 32) ----
  #pragma unroll
  for (int otile = 0; otile < 4; ++otile) {
    // Build both K-step A fragments (W3 rows) once per otile.
    // A layout: per-lane K split {lhalf*8 + 0..7, lhalf*8 + 16..23} (+ks*32)
    v16bf a0, a1;
    {
      int orow = otile * 16 + l16;
      const float* wrow = W3 + (size_t)orow * CIN + lhalf * 8;
      #pragma unroll
      for (int p = 0; p < 4; ++p) {
        f32x2 w0 = *(const f32x2*)(wrow + 2 * p);        // ks0, K = lhalf*8 + 2p
        f32x2 w1 = *(const f32x2*)(wrow + 16 + 2 * p);   // ks0, K = +16
        f32x2 w2 = *(const f32x2*)(wrow + 32 + 2 * p);   // ks1
        f32x2 w3 = *(const f32x2*)(wrow + 48 + 2 * p);   // ks1, K = +16
        a0[2 * p]     = (__bf16)w0.x;  a0[2 * p + 1]     = (__bf16)w0.y;
        a0[8 + 2 * p] = (__bf16)w1.x;  a0[8 + 2 * p + 1] = (__bf16)w1.y;
        a1[2 * p]     = (__bf16)w2.x;  a1[2 * p + 1]     = (__bf16)w2.y;
        a1[8 + 2 * p] = (__bf16)w3.x;  a1[8 + 2 * p + 1] = (__bf16)w3.y;
      }
    }
    for (int ctile = wave; ctile < NCT; ctile += 8) {
      int col = ctile * 16 + l16;
      // B fragments: N = col, K = ks*32 + {lhalf*8 + 0..7, lhalf*8 + 16..23} in Xb row
      const __bf16* xrow = Xb + (size_t)col * XB_STRIDE + lhalf * 8;
      bf16x8 b0l = *(const bf16x8*)(xrow);
      bf16x8 b0h = *(const bf16x8*)(xrow + 16);
      bf16x8 b1l = *(const bf16x8*)(xrow + 32);
      bf16x8 b1h = *(const bf16x8*)(xrow + 48);
      v16bf b0, b1;
      #pragma unroll
      for (int e = 0; e < 8; ++e) {
        b0[e] = b0l[e]; b0[8 + e] = b0h[e];
        b1[e] = b1l[e]; b1[8 + e] = b1h[e];
      }
      v8f acc = {};
      acc = __builtin_amdgcn_wmma_f32_16x16x32_bf16(false, a0, false, b0,
                                                    (short)0, acc, false, false);
      acc = __builtin_amdgcn_wmma_f32_16x16x32_bf16(false, a1, false, b1,
                                                    (short)0, acc, false, false);
      // D layout: VGPR j -> M=o row j (lanes<16) / j+8 (lanes>=16), N=col=lane%16
      int t = col / V_;
      int v = col % V_;
      int flane = t + 16 * (v >> 4);                 // fragment lane
      int fe    = v & 15;                            // fragment element
      #pragma unroll
      for (int j = 0; j < 8; ++j) {
        int o = otile * 16 + j + lhalf * 8;
        X3[(size_t)o * (TT * VP) + flane * 16 + fe] = (__bf16)(acc[j] + b3[o]);
      }
    }
  }
  __syncthreads();

  // ---- GEMM2: z[t][u] = sum_v x3[t][v] * adj[u][v], single K=32 WMMA ----
  const __bf16* an = adjb + (size_t)n * COUT * VP * VP;
  float* zn = z + (size_t)n * COUT * T_ * V_;
  for (int job = wave; job < COUT * 2; job += 8) {
    int o  = job >> 1;
    int uh = job & 1;
    // A fragment: contiguous per-lane 16 bf16 (stored pre-swizzled), 32B aligned
    v16bf a = *(const v16bf*)(X3 + (size_t)o * (TT * VP) + lane * 16);
    // B fragment = adj^T: N = u = uh*16 + lane%16 row, K = v = lhalf*16 + (0..15)
    // rows are 64B apart, offset 0/32B -> 32B aligned
    v16bf b = *(const v16bf*)(an + ((size_t)o * VP + uh * 16 + l16) * VP + lhalf * 16);
    v8f c = {};
    c = __builtin_amdgcn_wmma_f32_16x16x32_bf16(false, a, false, b,
                                                (short)0, c, false, false);
    int u = uh * 16 + l16;
    if (u < V_) {
      #pragma unroll
      for (int j = 0; j < 8; ++j) {
        int t = t0 + j + lhalf * 8;
        zn[((size_t)o * T_ + t) * V_ + u] = c[j];
      }
    }
  }
}

// ---------------- launcher ----------------
extern "C" void kernel_launch(void* const* d_in, const int* in_sizes, int n_in,
                              void* d_out, int out_size, void* d_ws, size_t ws_size,
                              hipStream_t stream) {
  const float* x  = (const float*)d_in[0];
  const float* A  = (const float*)d_in[1];
  const float* W1 = (const float*)d_in[2];
  const float* b1 = (const float*)d_in[3];
  const float* W2 = (const float*)d_in[4];
  const float* b2 = (const float*)d_in[5];
  const float* W3 = (const float*)d_in[6];
  const float* b3 = (const float*)d_in[7];
  const float* W4 = (const float*)d_in[8];
  const float* b4 = (const float*)d_in[9];
  float* z = (float*)d_out;

  // workspace: adjb (16 MB bf16, padded) then s (0.82 MB f32)
  __bf16* adjb = (__bf16*)d_ws;
  float*  s    = (float*)((char*)d_ws + (size_t)N_ * COUT * VP * VP * 2);

  k_sum_t<<<N_ * CIN, 256, 0, stream>>>(x, s);
  k_adj<<<N_, 256, 0, stream>>>(s, A, W1, b1, W2, b2, W4, b4, adjb);

  size_t lds_bytes = (size_t)COLS * XB_STRIDE * 2 + (size_t)COUT * TT * VP * 2; // ~120 KB
  k_main<<<N_ * (T_ / TT), 256, lds_bytes, stream>>>(x, W3, b3, adjb, z);
}
